// Dilated_res_block_57509612094126
// MI455X (gfx1250) — matrix operand order repaired and checked by reading, not compiled, not run on hardware
//
#include <hip/hip_runtime.h>
#include <hip/hip_bf16.h>

typedef __attribute__((ext_vector_type(16))) _Float16 v16h;
typedef __attribute__((ext_vector_type(8)))  float    v8f;

#define NPTS 16384
#define NB   2
#define KNB  16
#define EPS  1e-5f

// ---- f16 weight region offsets (in halves) inside d_ws ----
constexpr int OFF_FC    = 0;                       // fc_W      [32][32]
constexpr int OFF_M0    = OFF_FC    + 32 * 32;     // mlp[0].W  [64][32]
constexpr int OFF_M1    = OFF_M0    + 64 * 32;     // mlp[1].W  [64][64]
constexpr int OFF_M2    = OFF_M1    + 64 * 64;     // mlp[2].W  [64][64]
constexpr int OFF_M3    = OFF_M2    + 64 * 64;     // mlp[3].W  [128][64]
constexpr int OFF_LIN   = OFF_M3    + 128 * 64;    // lin_W     [128][2048]
constexpr int OFF_MLP2  = OFF_LIN   + 128 * 2048;  // mlp2.W    [128][128]
constexpr int OFF_SHORT = OFF_MLP2  + 128 * 128;   // short.W   [128][32]
constexpr int OFF_MLP1  = OFF_SHORT + 128 * 32;    // mlp1.W    [32][32]
constexpr int W16_TOTAL = OFF_MLP1  + 32 * 32;

constexpr size_t PC_OFF  = ((size_t)W16_TOTAL * 2 + 255) & ~(size_t)255;  // pc f32 [B*N][32]
constexpr size_t WT_OFF  = PC_OFF + (size_t)NB * NPTS * 32 * 4;           // wt f16 [B*N][16][16]
constexpr size_t SCL_OFF = WT_OFF + (size_t)NB * NPTS * 256 * 2;          // folded BN scales (f32)

// ---- per-channel folded scale/shift offsets (floats; sh array at +S_TOT) ----
constexpr int S_M0 = 0, S_M1 = 64, S_M2 = 128, S_M3 = 192;   // mlp[0..3]
constexpr int S_BL = 320, S_M2F = 448, S_SH = 576;           // bn_lin, mlp2, short
constexpr int S_M1P = 704, S_WN0 = 736, S_WN1 = 744, S_WN2 = 752;
constexpr int S_TOT = 768;

// ---- LDS carve (bytes) for the fused kernel ----
constexpr int L_AB = 0;            // A act buf  [256][<=128] f16 (64 KB)
constexpr int L_BB = 65536;        // B act buf  [256][<=64] f16 / NF chunk (32 KB)
constexpr int L_WT = 98304;        // wt tiles   [16][16][32] f16 (16 KB)
constexpr int L_X2 = 114688;       // bn_lin out [16][128] f16 (4 KB)
constexpr int L_FF = 118784;       // feature    [16][32]  f16 (1 KB)
constexpr int L_SC = 119808;       // shortcut   [16][128] f32 (8 KB)
constexpr int L_TOTAL = 131072;

// ---------------- WMMA fragment helpers (wave32, 16x16x32 f16) ----------------
// A (16x32, row per lane%16): lane<16 holds K{0..15}, lane>=16 K{16..31}
__device__ __forceinline__ v16h fragA(const _Float16* base, int ld, int row, int k0, int lane) {
    const _Float16* p = base + row * ld + k0 + (lane & 16);
    v16h f;
#pragma unroll
    for (int j = 0; j < 16; ++j) f[j] = p[j];
    return f;
}
// B (32x16, col per lane%16): lane<16 holds K{0..15}, lane>=16 K{16..31}.
// Activations stored X[col][c] -> one contiguous 32B load.
__device__ __forceinline__ v16h fragB(const _Float16* base, int ld, int col, int k0, int lane) {
    const _Float16* p = base + col * ld + k0 + (lane & 16);
    v16h f;
#pragma unroll
    for (int j = 0; j < 16; ++j) f[j] = p[j];
    return f;
}
__device__ __forceinline__ v8f wmma16(v16h a, v16h b, v8f c) {
    return __builtin_amdgcn_wmma_f32_16x16x32_f16(false, a, false, b, (short)0, c, false, false);
}

// ---------------- prep kernels ----------------
__global__ void k_cvt(const float* __restrict__ src, _Float16* __restrict__ dst, int n) {
    int i = blockIdx.x * blockDim.x + threadIdx.x;
    if (i < n) dst[i] = (_Float16)src[i];
}
// sc = g*rsqrt(v+eps); sh = ((cb + extra) - m)*sc + b   (cb/extra nullable)
__global__ void k_mkscale(const float* __restrict__ g, const float* __restrict__ b,
                          const float* __restrict__ m, const float* __restrict__ v,
                          const float* __restrict__ cb, const float* __restrict__ extra,
                          float* __restrict__ sc, float* __restrict__ sh, int n) {
    int i = blockIdx.x * blockDim.x + threadIdx.x;
    if (i >= n) return;
    float s = g[i] * rsqrtf(v[i] + EPS);
    float base = (cb ? cb[i] : 0.f) + (extra ? extra[i] : 0.f);
    sc[i] = s;
    sh[i] = (base - m[i]) * s + b[i];
}

// ---------------- mlp1: pc[b*N+n][c] = relu(BN(W @ feature[b,:,n])) ----------------
__global__ void k_mlp1(const float* __restrict__ feat, const _Float16* __restrict__ wsW,
                       const float* __restrict__ scl, float* __restrict__ pc) {
    int wid  = (blockIdx.x * blockDim.x + threadIdx.x) >> 5;  // one wave = 16 points
    int lane = threadIdx.x & 31;
    int gp0  = wid * 16;
    int b    = gp0 >> 14;
    int n    = (gp0 & (NPTS - 1)) + (lane & 15);
    const float* sc = scl + S_M1P;
    const float* sh = scl + S_TOT + S_M1P;
    v16h bf;
    int coff = lane & 16;
#pragma unroll
    for (int j = 0; j < 16; ++j)
        bf[j] = (_Float16)feat[((size_t)b * 32 + coff + j) * NPTS + n];
    const _Float16* W = wsW + OFF_MLP1;
    float* dst = pc + ((size_t)b * NPTS + n) * 32;
#pragma unroll
    for (int mt = 0; mt < 2; ++mt) {
        v16h a  = fragA(W, 32, mt * 16 + (lane & 15), 0, lane);
        v8f acc = {};
        acc = wmma16(a, bf, acc);
        int r0 = mt * 16 + ((lane & 16) ? 8 : 0);
#pragma unroll
        for (int i = 0; i < 8; ++i) {
            int r = r0 + i;
            float x = fmaf(acc[i], sc[r], sh[r]);
            dst[r] = x > 0.f ? x : 0.f;
        }
    }
}

// ---------------- weightnet: 3 -> 8 -> 8 -> 16 per (b,n,k) ----------------
__global__ void k_wn(const float* __restrict__ xyz, const int* __restrict__ neigh,
                     const float* __restrict__ W0, const float* __restrict__ W1,
                     const float* __restrict__ W2, const float* __restrict__ scl,
                     _Float16* __restrict__ wt) {
    int t = blockIdx.x * blockDim.x + threadIdx.x;
    if (t >= NB * NPTS * KNB) return;
    int k  = t & 15;
    int gp = t >> 4;                 // b*N + n
    int b  = gp >> 14;
    int idx = neigh[t];
    const float* sc0 = scl + S_WN0; const float* sh0 = scl + S_TOT + S_WN0;
    const float* sc1 = scl + S_WN1; const float* sh1 = scl + S_TOT + S_WN1;
    const float* sc2 = scl + S_WN2; const float* sh2 = scl + S_TOT + S_WN2;
    const float* xa = xyz + (size_t)gp * 3;
    const float* xb = xyz + ((size_t)b * NPTS + idx) * 3;
    float h0[3] = { xa[0] - xb[0], xa[1] - xb[1], xa[2] - xb[2] };
    float h1[8], h2[8];
#pragma unroll
    for (int o = 0; o < 8; ++o) {
        float y = W0[o * 3] * h0[0] + W0[o * 3 + 1] * h0[1] + W0[o * 3 + 2] * h0[2];
        y = fmaf(y, sc0[o], sh0[o]);
        h1[o] = y > 0.f ? y : 0.f;
    }
#pragma unroll
    for (int o = 0; o < 8; ++o) {
        float y = 0.f;
#pragma unroll
        for (int i = 0; i < 8; ++i) y += W1[o * 8 + i] * h1[i];
        y = fmaf(y, sc1[o], sh1[o]);
        h2[o] = y > 0.f ? y : 0.f;
    }
    _Float16* dst = wt + (size_t)gp * 256 + k;   // wt[gp][w][k]
#pragma unroll
    for (int o = 0; o < 16; ++o) {
        float y = 0.f;
#pragma unroll
        for (int i = 0; i < 8; ++i) y += W2[o * 8 + i] * h2[i];
        y = fmaf(y, sc2[o], sh2[o]);
        dst[o * 16] = (_Float16)(y > 0.f ? y : 0.f);
    }
}

// ---------------- fused main block ----------------
// GEMM layer over LDS activations: Xout[col][Cout] = epi(W @ Xin[col][Cin])
template <bool EPI>
__device__ void conv_layer(const _Float16* __restrict__ W, int Cin, int Cout,
                           const _Float16* Xin, int ldin, _Float16* Xout, int ldout,
                           const float* sc, const float* sh, int wave, int lane) {
    int mtiles = Cout >> 4;
    for (int t = wave; t < mtiles * 16; t += 8) {
        int mt = t % mtiles, nt = t / mtiles;
        v8f acc = {};
        for (int kc = 0; kc < Cin; kc += 32) {
            v16h a = fragA(W, Cin, mt * 16 + (lane & 15), kc, lane);
            v16h b = fragB(Xin, ldin, nt * 16 + (lane & 15), kc, lane);
            acc = wmma16(a, b, acc);
        }
        int col = nt * 16 + (lane & 15);
        int r0  = mt * 16 + ((lane & 16) ? 8 : 0);
        _Float16* dst = Xout + col * ldout + r0;
#pragma unroll
        for (int i = 0; i < 8; ++i) {
            float x = acc[i];
            if (EPI) {
                int r = r0 + i;
                x = fmaf(x, sc[r], sh[r]);
                x = x > 0.f ? x : 0.f;
            }
            dst[i] = (_Float16)x;
        }
    }
}

__global__ void __launch_bounds__(256)
k_fused(const float* __restrict__ pc, const _Float16* __restrict__ wt16,
        const float* __restrict__ feat, const int* __restrict__ neigh,
        const _Float16* __restrict__ wsW, const float* __restrict__ scl,
        float* __restrict__ out) {
    extern __shared__ char smem[];
    _Float16* Ab  = (_Float16*)(smem + L_AB);
    _Float16* Bb  = (_Float16*)(smem + L_BB);
    _Float16* wtL = (_Float16*)(smem + L_WT);
    _Float16* X2  = (_Float16*)(smem + L_X2);
    _Float16* Ff  = (_Float16*)(smem + L_FF);
    float*    SCm = (float*)   (smem + L_SC);

    int tid = threadIdx.x, wave = tid >> 5, lane = tid & 31;
    int gp0 = blockIdx.x * 16;          // 16 points per block
    int b   = gp0 >> 14;
    int n0  = gp0 & (NPTS - 1);

    // ---- stage fr0 = pc[n] - pc[idx]  -> Ab[col][32] ----
    {
        int col = tid;                  // col = p*16 + k
        int idx = neigh[gp0 * 16 + col];
        const float* pa = pc + (size_t)(gp0 + (col >> 4)) * 32;
        const float* pg = pc + ((size_t)b * NPTS + idx) * 32;
        _Float16* d = Ab + col * 32;
#pragma unroll
        for (int c = 0; c < 32; ++c) d[c] = (_Float16)(pa[c] - pg[c]);
    }
    // ---- stage wt tiles [p][w][32] (K padded to 32 with zeros) ----
    {
        int pnt = tid >> 4, w = tid & 15;
        const _Float16* src = wt16 + (size_t)(gp0 + pnt) * 256 + w * 16;
        _Float16* dst = wtL + (pnt * 16 + w) * 32;
#pragma unroll
        for (int k = 0; k < 16; ++k) dst[k] = src[k];
#pragma unroll
        for (int k = 16; k < 32; ++k) dst[k] = (_Float16)0.f;
    }
    // ---- stage feature columns [16][32] ----
    for (int t = tid; t < 512; t += 256) {
        int col = t >> 5, c = t & 31;
        Ff[col * 32 + c] = (_Float16)feat[((size_t)b * 32 + c) * NPTS + (n0 + col)];
    }
    __syncthreads();

    // ---- shortcut GEMM: SCm[col][128] = BN(short_W @ feature) ----
    {
        const float* sc = scl + S_SH; const float* sh = scl + S_TOT + S_SH;
        v16h a  = fragA(wsW + OFF_SHORT, 32, wave * 16 + (lane & 15), 0, lane);
        v16h bf = fragB(Ff, 32, (lane & 15), 0, lane);
        v8f acc = {};
        acc = wmma16(a, bf, acc);
        int col = lane & 15, r0 = wave * 16 + ((lane & 16) ? 8 : 0);
#pragma unroll
        for (int i = 0; i < 8; ++i) {
            int r = r0 + i;
            SCm[col * 128 + r] = fmaf(acc[i], sc[r], sh[r]);
        }
    }
    // ---- fc GEMM (raw): Bb[col][32] = fc_W @ fr0 ----
    conv_layer<false>(wsW + OFF_FC, 32, 32, Ab, 32, Bb, 32, nullptr, nullptr, wave, lane);
    __syncthreads();
    // ---- softmax over k, multiply fr0 in place (Ab) ----
    for (int t = tid; t < 512; t += 256) {
        int pnt = t >> 5, c = t & 31;
        float vals[16], mx = -1e30f;
#pragma unroll
        for (int k = 0; k < 16; ++k) { vals[k] = (float)Bb[(pnt * 16 + k) * 32 + c]; mx = fmaxf(mx, vals[k]); }
        float sm = 0.f;
#pragma unroll
        for (int k = 0; k < 16; ++k) { vals[k] = __expf(vals[k] - mx); sm += vals[k]; }
        float inv = 1.f / sm;
#pragma unroll
        for (int k = 0; k < 16; ++k) {
            int col = pnt * 16 + k;
            Ab[col * 32 + c] = (_Float16)((float)Ab[col * 32 + c] * vals[k] * inv);
        }
    }
    __syncthreads();
    // ---- mlp chain ----
    conv_layer<true>(wsW + OFF_M0, 32, 64, Ab, 32, Bb, 64,
                     scl + S_M0, scl + S_TOT + S_M0, wave, lane);
    __syncthreads();
    conv_layer<true>(wsW + OFF_M1, 64, 64, Bb, 64, Ab, 64,
                     scl + S_M1, scl + S_TOT + S_M1, wave, lane);
    __syncthreads();
    conv_layer<true>(wsW + OFF_M2, 64, 64, Ab, 64, Bb, 64,
                     scl + S_M2, scl + S_TOT + S_M2, wave, lane);
    __syncthreads();
    conv_layer<true>(wsW + OFF_M3, 64, 128, Bb, 64, Ab, 128,
                     scl + S_M3, scl + S_TOT + S_M3, wave, lane);

    // ---- fused einsum (nf^T[w][c] = wt @ fr3^T, K=16 padded) + linear GEMM ----
    v8f accL = {};
    _Float16* NFc = Bb;                 // rotating chunk: [16 pts][256 j] f16 (8 KB)
    for (int ct = 0; ct < 8; ++ct) {    // c-tile of fr3 channels
        __syncthreads();                // protect NFc reuse / fr3 stores
        for (int pp = wave; pp < 16; pp += 8) {
            v16h a = fragA(wtL + pp * 512, 32, (lane & 15), 0, lane);
            v16h bf = {};
            if (lane < 16) {            // lanes>=16 cover K=16..31 -> zero pad
                int c = ct * 16 + (lane & 15);
#pragma unroll
                for (int j = 0; j < 16; ++j) bf[j] = Ab[(pp * 16 + j) * 128 + c];
            }
            v8f d = {};
            d = wmma16(a, bf, d);
            // j = c*16 + w ; rows=w contiguous per lane
            _Float16* dst = NFc + pp * 256 + (lane & 15) * 16 + ((lane & 16) ? 8 : 0);
#pragma unroll
            for (int i = 0; i < 8; ++i) dst[i] = (_Float16)d[i];
        }
        __syncthreads();
        for (int cc = 0; cc < 8; ++cc) {  // 8 K-chunks of 32 inside this 256-j chunk
            v16h a = fragA(wsW + OFF_LIN, 2048, wave * 16 + (lane & 15), ct * 256 + cc * 32, lane);
            v16h bf = fragB(NFc, 256, (lane & 15), cc * 32, lane);
            accL = wmma16(a, bf, accL);
        }
    }
    // ---- bn_lin epilogue (lin_b folded into shift) -> X2[col][128] f16 ----
    {
        const float* sc = scl + S_BL; const float* sh = scl + S_TOT + S_BL;
        int col = lane & 15, r0 = wave * 16 + ((lane & 16) ? 8 : 0);
#pragma unroll
        for (int i = 0; i < 8; ++i) {
            int r = r0 + i;
            X2[col * 128 + r] = (_Float16)fmaf(accL[i], sc[r], sh[r]);
        }
    }
    __syncthreads();
    // ---- mlp2 GEMM + residual + leaky ReLU -> out[b][r][n] ----
    {
        const float* sc = scl + S_M2F; const float* sh = scl + S_TOT + S_M2F;
        v8f acc = {};
        for (int kc = 0; kc < 128; kc += 32) {
            v16h a  = fragA(wsW + OFF_MLP2, 128, wave * 16 + (lane & 15), kc, lane);
            v16h bf = fragB(X2, 128, (lane & 15), kc, lane);
            acc = wmma16(a, bf, acc);
        }
        int col = lane & 15, n = n0 + col;
        int r0 = wave * 16 + ((lane & 16) ? 8 : 0);
#pragma unroll
        for (int i = 0; i < 8; ++i) {
            int r = r0 + i;
            float x = fmaf(acc[i], sc[r], sh[r]) + SCm[col * 128 + r];
            out[((size_t)b * 128 + r) * NPTS + n] = x > 0.f ? x : 0.2f * x;
        }
    }
}

// ---------------- host ----------------
extern "C" void kernel_launch(void* const* d_in, const int* in_sizes, int n_in,
                              void* d_out, int out_size, void* d_ws, size_t ws_size,
                              hipStream_t stream) {
#define FP(i) ((const float*)d_in[i])
    const float* feat  = FP(0);
    const int*   neigh = (const int*)d_in[2];
    char* ws = (char*)d_ws;
    _Float16* wsW  = (_Float16*)ws;
    float*    pc   = (float*)(ws + PC_OFF);
    _Float16* wt16 = (_Float16*)(ws + WT_OFF);
    float*    scl  = (float*)(ws + SCL_OFF);

    auto cvt = [&](const float* src, int off, int n) {
        k_cvt<<<(n + 255) / 256, 256, 0, stream>>>(src, wsW + off, n);
    };
    // jax pytree (sorted-key) parameter index map — see round-0 analysis
    cvt(FP(7),  OFF_FC,    32 * 32);
    cvt(FP(10), OFF_M0,    64 * 32);
    cvt(FP(16), OFF_M1,    64 * 64);
    cvt(FP(22), OFF_M2,    64 * 64);
    cvt(FP(28), OFF_M3,    128 * 64);
    cvt(FP(8),  OFF_LIN,   128 * 2048);
    cvt(FP(39), OFF_MLP2,  128 * 128);
    cvt(FP(44), OFF_SHORT, 128 * 32);
    cvt(FP(34), OFF_MLP1,  32 * 32);

    auto mks = [&](const float* g, const float* b, const float* m, const float* v,
                   const float* cb, const float* ex, int soff, int n) {
        k_mkscale<<<1, ((n + 63) / 64) * 64, 0, stream>>>(g, b, m, v, cb, ex,
                                                          scl + soff, scl + S_TOT + soff, n);
    };
    mks(FP(13), FP(12), FP(14), FP(15), FP(11), nullptr, S_M0, 64);    // mlp[0]
    mks(FP(19), FP(18), FP(20), FP(21), FP(17), nullptr, S_M1, 64);    // mlp[1]
    mks(FP(25), FP(24), FP(26), FP(27), FP(23), nullptr, S_M2, 64);    // mlp[2]
    mks(FP(31), FP(30), FP(32), FP(33), FP(29), nullptr, S_M3, 128);   // mlp[3]
    mks(FP(4),  FP(3),  FP(5),  FP(6),  nullptr, FP(9),  S_BL, 128);   // bn_lin (+lin_b)
    mks(FP(41), FP(40), FP(42), FP(43), nullptr, nullptr, S_M2F, 128); // mlp2
    mks(FP(46), FP(45), FP(47), FP(48), nullptr, nullptr, S_SH, 128);  // short
    mks(FP(36), FP(35), FP(37), FP(38), nullptr, nullptr, S_M1P, 32);  // mlp1
    mks(FP(52), FP(51), FP(53), FP(54), FP(50), nullptr, S_WN0, 8);    // wn[0]
    mks(FP(58), FP(57), FP(59), FP(60), FP(56), nullptr, S_WN1, 8);    // wn[1]
    mks(FP(64), FP(63), FP(65), FP(66), FP(62), nullptr, S_WN2, 16);   // wn[2]

    // mlp1 -> pc  (2048 waves of 16 points)
    k_mlp1<<<256, 256, 0, stream>>>(feat, wsW, scl, pc);
    // weightnet -> wt16
    k_wn<<<(NB * NPTS * KNB + 255) / 256, 256, 0, stream>>>(FP(1), neigh,
                                                            FP(49), FP(55), FP(61), scl, wt16);
    // fused main pipeline
    (void)hipFuncSetAttribute((const void*)k_fused,
                              hipFuncAttributeMaxDynamicSharedMemorySize, L_TOTAL);
    k_fused<<<NB * NPTS / 16, 256, L_TOTAL, stream>>>(pc, wt16, feat, neigh, wsW, scl,
                                                      (float*)d_out);
#undef FP
}
